// Attention_m_17815524344494
// MI455X (gfx1250) — matrix operand order repaired and checked by reading, not compiled, not run on hardware
//
#include <hip/hip_runtime.h>
#include <hip/hip_bf16.h>

// MI455X (gfx1250) fused multi-head attention, wave32 + v_wmma_f32_16x16x32_f16.
// Pipeline: cvt X->f16; cvt+transpose W->f16 [out][in]; QKV proj (WMMA);
// flash attention (async global->LDS staging of K/V, WMMA + online softmax,
// probs reshaped via LDS); output proj (WMMA).
// Workspace use: X16 (24MB) + Wt (4.5MB) + Q/K/V/ctx f16 (4x24MB) ~= 131 MB.

typedef _Float16 half_t;
typedef __attribute__((ext_vector_type(8)))  _Float16 v8h;
typedef __attribute__((ext_vector_type(16))) _Float16 v16h;
typedef __attribute__((ext_vector_type(8)))  float    v8f;

#define H_DIM   768
#define NHEADS  12
#define HS      64
#define PP      512
#define NROWS   16384          // B*M*P = 8*4*512
#define WELEMS  (768*768)

// dynamic LDS layout for the attention kernel (in halves):
//   [0, PP*HS)            K tile   [p][hs]   64 KB
//   [PP*HS, 2*PP*HS)      V^T tile [hs][p]   64 KB
//   [2*PP*HS, +8*16*32)   per-wave probs staging, 8 KB
#define ATTN_LDS_HALVES (2 * PP * HS + 8 * 16 * 32)
#define ATTN_LDS_BYTES  (ATTN_LDS_HALVES * 2)

__device__ inline v16h cat8(v8h lo, v8h hi) {
  return __builtin_shufflevector(lo, hi, 0,1,2,3,4,5,6,7,8,9,10,11,12,13,14,15);
}

// A fragment (16x32 f16, MxK) from row-major [row][k], row stride ld halves.
// Lane l: M = l&15; lanes 0-15 hold K 0-15, lanes 16-31 hold K 16-31 (contiguous).
__device__ inline v16h load_a_frag(const half_t* base, int ld, int lane) {
  const half_t* r = base + (size_t)(lane & 15) * ld + ((lane & 16) ? 16 : 0);
  v8h lo = *(const v8h*)(r);
  v8h hi = *(const v8h*)(r + 8);
  return cat8(lo, hi);
}

// B fragment (32x16 f16, KxN) from a matrix stored [N][K] row-major (ld halves).
// Lane l: N = l&15; lanes 0-15 hold K 0-15, lanes 16-31 hold K 16-31 (contiguous).
__device__ inline v16h load_b_frag(const half_t* base, int ld, int lane) {
  const half_t* r = base + (size_t)(lane & 15) * ld + ((lane & 16) ? 16 : 0);
  v8h lo = *(const v8h*)(r);
  v8h hi = *(const v8h*)(r + 8);
  return cat8(lo, hi);
}

__device__ inline v8f wmma16(v16h a, v16h b, v8f c) {
  return __builtin_amdgcn_wmma_f32_16x16x32_f16(false, a, false, b, (short)0, c,
                                                false, false);
}

// Async 16-byte global -> LDS copy (per active lane). Tracked by ASYNCcnt.
__device__ inline void async_g2l_b128(unsigned lds_byte_off, const void* gaddr) {
  asm volatile("global_load_async_to_lds_b128 %0, %1, off"
               :: "v"(lds_byte_off), "v"(gaddr) : "memory");
}

// ---------------- prep kernels ----------------
__global__ void cvt_x_kernel(const float* __restrict__ x, half_t* __restrict__ y, int n) {
  int i = blockIdx.x * blockDim.x + threadIdx.x;
  if (i < n) y[i] = (half_t)x[i];
}

// Wt[m][out][in] = (f16) W[m][in][out]
__global__ void cvt_w_kernel(const float* __restrict__ Wq, const float* __restrict__ Wk,
                             const float* __restrict__ Wv, const float* __restrict__ Wo,
                             half_t* __restrict__ Wt) {
  int i = blockIdx.x * blockDim.x + threadIdx.x;
  if (i >= 4 * WELEMS) return;
  int m = i / WELEMS;
  int r = i - m * WELEMS;
  int row = r / H_DIM;           // input dim
  int col = r - row * H_DIM;     // output dim
  const float* W = (m == 0) ? Wq : (m == 1) ? Wk : (m == 2) ? Wv : Wo;
  Wt[(size_t)m * WELEMS + (size_t)col * H_DIM + row] = (half_t)W[(size_t)row * H_DIM + col];
}

// ---------------- QKV projection ----------------
// grid = (NROWS/16, NHEADS, 3); block = 32 (one wave per 16x64 tile)
__global__ __launch_bounds__(32)
void qkv_proj_kernel(const half_t* __restrict__ X16, const half_t* __restrict__ Wt,
                     const float* __restrict__ bq, const float* __restrict__ bk,
                     const float* __restrict__ bv,
                     half_t* __restrict__ Qo, half_t* __restrict__ Ko,
                     half_t* __restrict__ Vo) {
  int tm = blockIdx.x, head = blockIdx.y, w = blockIdx.z;
  int lane = threadIdx.x;
  const half_t* Wm = Wt + (size_t)w * WELEMS;
  const float* bias = (w == 0) ? bq : (w == 1) ? bk : bv;

  v8f acc[4];
  for (int t = 0; t < 4; ++t) for (int j = 0; j < 8; ++j) acc[t][j] = 0.0f;

  const half_t* Arow = X16 + (size_t)tm * 16 * H_DIM;
  for (int k0 = 0; k0 < H_DIM; k0 += 32) {
    v16h a = load_a_frag(Arow + k0, H_DIM, lane);
    #pragma unroll
    for (int t = 0; t < 4; ++t) {
      v16h b = load_b_frag(Wm + (size_t)(head * HS + t * 16) * H_DIM + k0, H_DIM, lane);
      acc[t] = wmma16(a, b, acc[t]);
    }
  }

  int bm = tm >> 5;                 // 32 row-tiles per (b,m) slab
  int p0 = (tm & 31) << 4;
  int n = lane & 15, rbase = (lane >> 4) << 3;
  size_t hdbase = ((size_t)bm * NHEADS + head);
  #pragma unroll
  for (int t = 0; t < 4; ++t) {
    int hs = t * 16 + n;
    float b = bias[head * HS + hs];
    #pragma unroll
    for (int j = 0; j < 8; ++j) {
      int p = p0 + rbase + j;
      float val = acc[t][j] + b;
      if (w == 0)       Qo[(hdbase * PP + p) * HS + hs] = (half_t)(val * 0.125f);
      else if (w == 1)  Ko[(hdbase * PP + p) * HS + hs] = (half_t)val;
      else              Vo[(hdbase * HS + hs) * PP + p] = (half_t)val;  // V transposed
    }
  }
}

// ---------------- flash attention ----------------
// grid = (32 /*b*m*/, NHEADS); block = 256 (8 waves).
// K [512][64] and V^T [64][512] for this head are async-staged into LDS once,
// then all 8 waves stream their score/PV B-fragments from LDS.
__global__ __launch_bounds__(256)
void attn_kernel(const half_t* __restrict__ Q, const half_t* __restrict__ K,
                 const half_t* __restrict__ V, half_t* __restrict__ C) {
  extern __shared__ __align__(16) half_t dynlds[];
  int bm = blockIdx.x, head = blockIdx.y;
  int tid = threadIdx.x;
  int wave = tid >> 5, lane = tid & 31;
  size_t hdbase = (size_t)bm * NHEADS + head;
  const half_t* Qb = Q + hdbase * PP * HS;   // [p][hs]
  const half_t* Kb = K + hdbase * PP * HS;   // [p][hs]
  const half_t* Vb = V + hdbase * HS * PP;   // [hs][p]  (transposed)

  // ---- async stage K (64KB) + V^T (64KB) into LDS ----
  unsigned lds_base = (unsigned)(uintptr_t)dynlds;   // low 32 bits = LDS byte addr
  const unsigned BUF_BYTES = PP * HS * 2;            // 65536 per buffer
  for (int i = tid; i < 2 * PP * HS / 8; i += 256) { // 8192 x 16B transfers
    unsigned boff = (unsigned)i * 16u;
    if (boff < BUF_BYTES)
      async_g2l_b128(lds_base + boff, (const char*)Kb + boff);
    else
      async_g2l_b128(lds_base + boff, (const char*)Vb + (boff - BUF_BYTES));
  }
  asm volatile("s_wait_asynccnt 0x0" ::: "memory");
  __syncthreads();

  const half_t* kl = dynlds;                 // [p][hs] in LDS
  const half_t* vl = dynlds + PP * HS;       // [hs][p] in LDS
  half_t* myl = dynlds + 2 * PP * HS + wave * (16 * 32);

  for (int qt = wave; qt < PP / 16; qt += 8) {
    v16h a0 = load_a_frag(Qb + (size_t)qt * 16 * HS, HS, lane);       // hs 0..31
    v16h a1 = load_a_frag(Qb + (size_t)qt * 16 * HS + 32, HS, lane);  // hs 32..63

    float mrow[8], lrow[8];
    v8f o[4];
    #pragma unroll
    for (int j = 0; j < 8; ++j) { mrow[j] = -1e30f; lrow[j] = 0.0f; }
    for (int t = 0; t < 4; ++t) for (int j = 0; j < 8; ++j) o[t][j] = 0.0f;

    for (int kc = 0; kc < PP; kc += 32) {
      // scores for keys [kc, kc+32): two 16x16 f32 D-fragments (K from LDS)
      v8f s0, s1;
      for (int j = 0; j < 8; ++j) { s0[j] = 0.0f; s1[j] = 0.0f; }
      s0 = wmma16(a0, load_b_frag(kl + (size_t)kc * HS,             HS, lane), s0);
      s0 = wmma16(a1, load_b_frag(kl + (size_t)kc * HS + 32,        HS, lane), s0);
      s1 = wmma16(a0, load_b_frag(kl + (size_t)(kc + 16) * HS,      HS, lane), s1);
      s1 = wmma16(a1, load_b_frag(kl + (size_t)(kc + 16) * HS + 32, HS, lane), s1);

      // online softmax: each lane holds 8 query rows (j), row spread across 16 lanes
      float mnew[8], sc[8], rs[8], e0[8], e1[8];
      #pragma unroll
      for (int j = 0; j < 8; ++j) mnew[j] = fmaxf(mrow[j], fmaxf(s0[j], s1[j]));
      #pragma unroll
      for (int mask = 1; mask < 16; mask <<= 1)
        #pragma unroll
        for (int j = 0; j < 8; ++j)
          mnew[j] = fmaxf(mnew[j], __shfl_xor(mnew[j], mask, 32));
      #pragma unroll
      for (int j = 0; j < 8; ++j) {
        e0[j] = __expf(s0[j] - mnew[j]);
        e1[j] = __expf(s1[j] - mnew[j]);
        sc[j] = __expf(mrow[j] - mnew[j]);
        rs[j] = e0[j] + e1[j];
        mrow[j] = mnew[j];
      }
      #pragma unroll
      for (int mask = 1; mask < 16; mask <<= 1)
        #pragma unroll
        for (int j = 0; j < 8; ++j)
          rs[j] += __shfl_xor(rs[j], mask, 32);
      #pragma unroll
      for (int j = 0; j < 8; ++j) lrow[j] = lrow[j] * sc[j] + rs[j];
      for (int t = 0; t < 4; ++t)
        #pragma unroll
        for (int j = 0; j < 8; ++j) o[t][j] *= sc[j];

      // D-fragment -> A-fragment reshape via LDS (wave-private 16x32 f16 tile)
      int n = lane & 15, rbase = (lane >> 4) << 3;
      #pragma unroll
      for (int j = 0; j < 8; ++j) {
        myl[(rbase + j) * 32 + n]      = (half_t)e0[j];
        myl[(rbase + j) * 32 + 16 + n] = (half_t)e1[j];
      }
      asm volatile("s_wait_dscnt 0" ::: "memory");   // same-wave LDS RAW fence
      v16h pa = load_a_frag(myl, 32, lane);          // probs 16x32 A-fragment
      #pragma unroll
      for (int t = 0; t < 4; ++t)
        o[t] = wmma16(pa, load_b_frag(vl + (size_t)(t * 16) * PP + kc, PP, lane), o[t]);
    }

    // normalize and store ctx (f16, merged-head row-major [token][H])
    int n = lane & 15, rbase = (lane >> 4) << 3;
    size_t row0 = (size_t)bm * PP + (size_t)qt * 16;
    float inv[8];
    #pragma unroll
    for (int j = 0; j < 8; ++j) inv[j] = 1.0f / lrow[j];
    #pragma unroll
    for (int t = 0; t < 4; ++t) {
      int col = head * HS + t * 16 + n;
      #pragma unroll
      for (int j = 0; j < 8; ++j)
        C[(row0 + rbase + j) * H_DIM + col] = (half_t)(o[t][j] * inv[j]);
    }
  }
}

// ---------------- output projection ----------------
// grid = (NROWS/16, H/64); block = 32
__global__ __launch_bounds__(32)
void oproj_kernel(const half_t* __restrict__ Cx, const half_t* __restrict__ Wot,
                  const float* __restrict__ bo, float* __restrict__ out) {
  int tm = blockIdx.x, ns = blockIdx.y, lane = threadIdx.x;
  v8f acc[4];
  for (int t = 0; t < 4; ++t) for (int j = 0; j < 8; ++j) acc[t][j] = 0.0f;

  const half_t* Arow = Cx + (size_t)tm * 16 * H_DIM;
  for (int k0 = 0; k0 < H_DIM; k0 += 32) {
    v16h a = load_a_frag(Arow + k0, H_DIM, lane);
    #pragma unroll
    for (int t = 0; t < 4; ++t) {
      v16h b = load_b_frag(Wot + (size_t)(ns * 64 + t * 16) * H_DIM + k0, H_DIM, lane);
      acc[t] = wmma16(a, b, acc[t]);
    }
  }
  int n = lane & 15, rbase = (lane >> 4) << 3;
  #pragma unroll
  for (int t = 0; t < 4; ++t) {
    int col = ns * 64 + t * 16 + n;
    float b = bo[col];
    #pragma unroll
    for (int j = 0; j < 8; ++j)
      out[((size_t)tm * 16 + rbase + j) * H_DIM + col] = acc[t][j] + b;
  }
}

extern "C" void kernel_launch(void* const* d_in, const int* in_sizes, int n_in,
                              void* d_out, int out_size, void* d_ws, size_t ws_size,
                              hipStream_t stream) {
  const float* hsx = (const float*)d_in[0];
  const float* Wq  = (const float*)d_in[1];
  const float* bq  = (const float*)d_in[2];
  const float* Wk  = (const float*)d_in[3];
  const float* bk  = (const float*)d_in[4];
  const float* Wv  = (const float*)d_in[5];
  const float* bv  = (const float*)d_in[6];
  const float* Wo  = (const float*)d_in[7];
  const float* bo  = (const float*)d_in[8];

  half_t* ws = (half_t*)d_ws;
  const size_t NX  = (size_t)NROWS * H_DIM;     // halves per activation buffer
  const size_t NW  = 4ull * WELEMS;
  size_t offX = 0;
  size_t offW = offX + NX;
  size_t offQ = offW + NW;
  size_t offK = offQ + NX;
  size_t offV = offK + NX;
  size_t offC = offV + NX;

  cvt_x_kernel<<<(unsigned)((NX + 255) / 256), 256, 0, stream>>>(hsx, ws + offX, (int)NX);
  cvt_w_kernel<<<(unsigned)((NW + 255) / 256), 256, 0, stream>>>(Wq, Wk, Wv, Wo, ws + offW);

  qkv_proj_kernel<<<dim3(NROWS / 16, NHEADS, 3), 32, 0, stream>>>(
      ws + offX, ws + offW, bq, bk, bv, ws + offQ, ws + offK, ws + offV);

  attn_kernel<<<dim3(NROWS / PP, NHEADS), 256, ATTN_LDS_BYTES, stream>>>(
      ws + offQ, ws + offK, ws + offV, ws + offC);

  oproj_kernel<<<dim3(NROWS / 16, H_DIM / 64), 32, 0, stream>>>(
      ws + offC, ws + offW + 3ull * WELEMS, bo, (float*)d_out);
}
